// NodeModel_85478439125101
// MI455X (gfx1250) — matrix operand order repaired and checked by reading, not compiled, not run on hardware
//
#include <hip/hip_runtime.h>
#include <hip/hip_bf16.h>

#define N_NODES 100000
#define N_EDGES 1600000
#define D_NODE  128
#define K1PAD   288      // 259 padded up to 9*32
#define KT1     9        // K-tiles layer 1
#define KT2     4        // K-tiles layer 2 (128/32)
#define NT      8        // N-tiles (128/16)
#define BM      32       // node rows per block
#define AS      296      // Ash row stride in halves (multiple of 8)
#define HS      136      // Hsh row stride in halves (multiple of 8)

typedef _Float16 v16h __attribute__((ext_vector_type(16)));
typedef _Float16 v8h  __attribute__((ext_vector_type(8)));
typedef float    v8f  __attribute__((ext_vector_type(8)));

// ---------------------------------------------------------------------------
// 1) zero the per-node accumulator: {sum_attr0, sum_attr1, sum_attr2, cnt}
// ---------------------------------------------------------------------------
__global__ void zero_agg_kernel(float* __restrict__ agg) {
    int i = blockIdx.x * 256 + threadIdx.x;
    if (i < N_NODES * 4) agg[i] = 0.0f;
}

// ---------------------------------------------------------------------------
// 2) edge scatter: only edge_attr (+count) needs real scatter; the 128-dim
//    part of the scatter-mean collapses to x[n] * (cnt[n] > 0).
// ---------------------------------------------------------------------------
__global__ void scatter_edges_kernel(const int* __restrict__ edge_index,
                                     const float* __restrict__ edge_attr,
                                     float* __restrict__ agg) {
    int e = blockIdx.x * 256 + threadIdx.x;
    if (e >= N_EDGES) return;
    int src = edge_index[N_EDGES + e];          // edge_index[1][e]
    float a0 = edge_attr[3 * e + 0];
    float a1 = edge_attr[3 * e + 1];
    float a2 = edge_attr[3 * e + 2];
    float* p = agg + (size_t)src * 4;
    unsafeAtomicAdd(p + 0, a0);
    unsafeAtomicAdd(p + 1, a1);
    unsafeAtomicAdd(p + 2, a2);
    unsafeAtomicAdd(p + 3, 1.0f);
}

// ---------------------------------------------------------------------------
// 3) pack W1 (zero-padded to 288 rows) and W2 into f16 WMMA-B fragment order:
//    frag[((kt*NT + nt)*32 + lane)*16 + e] = W[kt*32 + 16*(lane>>4) + e][nt*16 + lane&15]
//    so each lane's B fragment is one contiguous 32-byte load.
// ---------------------------------------------------------------------------
__global__ void pack_weights_kernel(const float* __restrict__ W1,
                                    const float* __restrict__ W2,
                                    _Float16* __restrict__ W1f,
                                    _Float16* __restrict__ W2f) {
    int idx = blockIdx.x * 256 + threadIdx.x;
    const int tot1 = KT1 * NT * 32 * 16;     // 36864
    const int tot2 = KT2 * NT * 32 * 16;     // 16384
    if (idx < tot1) {
        int e  = idx & 15;
        int L  = (idx >> 4) & 31;
        int nt = (idx >> 9) & 7;
        int kt = idx >> 12;
        int k  = kt * 32 + 16 * (L >> 4) + e;
        int n  = nt * 16 + (L & 15);
        W1f[idx] = (k < 259) ? (_Float16)W1[k * 128 + n] : (_Float16)0.0f;
    } else if (idx < tot1 + tot2) {
        int j  = idx - tot1;
        int e  = j & 15;
        int L  = (j >> 4) & 31;
        int nt = (j >> 9) & 7;
        int kt = j >> 12;
        int k  = kt * 32 + 16 * (L >> 4) + e;
        int n  = nt * 16 + (L & 15);
        W2f[j] = (_Float16)W2[k * 128 + n];
    }
}

// ---------------------------------------------------------------------------
// helper: load A-fragment (16x32 f16) from LDS row-major tile.
// lane L holds row m0+(L&15); K = k0+16*hi+e
// => two contiguous 16-byte LDS loads per lane.
// ---------------------------------------------------------------------------
__device__ __forceinline__ v16h load_afrag(const _Float16* rowptr, int k0, int hi) {
    const _Float16* p = rowptr + k0 + 16 * hi;
    v8h lo = *(const v8h*)(p);
    v8h up = *(const v8h*)(p + 8);
    v16h a;
#pragma unroll
    for (int i = 0; i < 8; ++i) { a[i] = lo[i]; a[i + 8] = up[i]; }
    return a;
}

// ---------------------------------------------------------------------------
// 4) fused MLP: 32 nodes/block, 8 waves = 8 N-tiles, 2 M-subtiles per wave.
// ---------------------------------------------------------------------------
__global__ void __launch_bounds__(256)
mlp_kernel(const float* __restrict__ x,
           const float* __restrict__ agg,
           const _Float16* __restrict__ W1f,
           const _Float16* __restrict__ W2f,
           const float* __restrict__ b1,
           const float* __restrict__ b2,
           float* __restrict__ out) {
    __shared__ __align__(16) _Float16 Ash[BM * AS];   // [32][296] input tile (K padded)
    __shared__ __align__(16) _Float16 Hsh[BM * HS];   // [32][136] hidden tile
    __shared__ float sAgg[BM][3];
    __shared__ float sMask[BM];

    const int tid  = threadIdx.x;
    const int base = blockIdx.x * BM;

    // per-row aggregated edge data
    if (tid < BM) {
        const float4 g = ((const float4*)agg)[base + tid];
        float c   = g.w;
        float inv = 1.0f / fmaxf(c, 1.0f);
        sAgg[tid][0] = g.x * inv;
        sAgg[tid][1] = g.y * inv;
        sAgg[tid][2] = g.z * inv;
        sMask[tid]   = (c > 0.0f) ? 1.0f : 0.0f;
    }
    __syncthreads();

    // cols [0:128) = x, cols [128:256) = x * mask  (128 floats = 32 float4 / row)
#pragma unroll 2
    for (int i = tid; i < BM * 32; i += 256) {
        int row = i >> 5, c4 = i & 31;
        float4 xv = ((const float4*)x)[(size_t)(base + row) * 32 + c4];
        float m = sMask[row];
        _Float16* p = &Ash[row * AS + c4 * 4];
        p[0] = (_Float16)xv.x; p[1] = (_Float16)xv.y;
        p[2] = (_Float16)xv.z; p[3] = (_Float16)xv.w;
        _Float16* q = &Ash[row * AS + 128 + c4 * 4];
        q[0] = (_Float16)(xv.x * m); q[1] = (_Float16)(xv.y * m);
        q[2] = (_Float16)(xv.z * m); q[3] = (_Float16)(xv.w * m);
    }
    // zero pad cols [259:296)
    for (int i = tid; i < BM * 37; i += 256) {
        int row = i / 37, col = 259 + (i % 37);
        Ash[row * AS + col] = (_Float16)0.0f;
    }
    // cols [256:259) = mean edge_attr
    if (tid < BM * 3) {
        int row = tid / 3, j = tid - row * 3;
        Ash[row * AS + 256 + j] = (_Float16)sAgg[row][j];
    }
    __syncthreads();

    const int lane = tid & 31;
    const int nt   = tid >> 5;          // wave id = N-tile
    const int col  = lane & 15;
    const int hi   = lane >> 4;
    const int n    = nt * 16 + col;

    // ---------------- layer 1: [32 x 288] @ [288 x 128] ----------------
    v8f acc0 = {};
    v8f acc1 = {};
#pragma unroll
    for (int kt = 0; kt < KT1; ++kt) {
        v16h bfrag = *(const v16h*)(W1f + (((kt * NT + nt) * 32 + lane) << 4));
        int k0 = kt * 32;
        v16h a0 = load_afrag(&Ash[(col) * AS], k0, hi);
        v16h a1 = load_afrag(&Ash[(16 + col) * AS], k0, hi);
        acc0 = __builtin_amdgcn_wmma_f32_16x16x32_f16(false, a0, false, bfrag,
                                                      (short)0, acc0, false, false);
        acc1 = __builtin_amdgcn_wmma_f32_16x16x32_f16(false, a1, false, bfrag,
                                                      (short)0, acc1, false, false);
    }
    // bias + relu -> f16 hidden tile (C layout: lane col = N, VGPR r -> M = r + 8*hi)
    {
        float bb = b1[n];
#pragma unroll
        for (int r = 0; r < 8; ++r) {
            int m = r + 8 * hi;
            Hsh[m * HS + n]        = (_Float16)fmaxf(acc0[r] + bb, 0.0f);
            Hsh[(16 + m) * HS + n] = (_Float16)fmaxf(acc1[r] + bb, 0.0f);
        }
    }
    __syncthreads();

    // ---------------- layer 2: [32 x 128] @ [128 x 128] ----------------
    v8f o0 = {};
    v8f o1 = {};
#pragma unroll
    for (int kt = 0; kt < KT2; ++kt) {
        v16h bfrag = *(const v16h*)(W2f + (((kt * NT + nt) * 32 + lane) << 4));
        int k0 = kt * 32;
        v16h a0 = load_afrag(&Hsh[(col) * HS], k0, hi);
        v16h a1 = load_afrag(&Hsh[(16 + col) * HS], k0, hi);
        o0 = __builtin_amdgcn_wmma_f32_16x16x32_f16(false, a0, false, bfrag,
                                                    (short)0, o0, false, false);
        o1 = __builtin_amdgcn_wmma_f32_16x16x32_f16(false, a1, false, bfrag,
                                                    (short)0, o1, false, false);
    }
    {
        float bb = b2[n];
#pragma unroll
        for (int r = 0; r < 8; ++r) {
            int m = r + 8 * hi;
            out[(size_t)(base + m) * 128 + n]      = o0[r] + bb;
            out[(size_t)(base + 16 + m) * 128 + n] = o1[r] + bb;
        }
    }
}

// ---------------------------------------------------------------------------
extern "C" void kernel_launch(void* const* d_in, const int* in_sizes, int n_in,
                              void* d_out, int out_size, void* d_ws, size_t ws_size,
                              hipStream_t stream) {
    const float* x  = (const float*)d_in[0];
    const int*   ei = (const int*)d_in[1];
    const float* ea = (const float*)d_in[2];
    // d_in[3] = u, d_in[4] = batch : unused by the reference computation
    const float* W1 = (const float*)d_in[5];
    const float* b1 = (const float*)d_in[6];
    const float* W2 = (const float*)d_in[7];
    const float* b2 = (const float*)d_in[8];
    float* out = (float*)d_out;

    char* ws = (char*)d_ws;
    float*    agg = (float*)ws;                                   // N*4 floats
    _Float16* W1f = (_Float16*)(ws + (size_t)N_NODES * 4 * sizeof(float));
    _Float16* W2f = W1f + (size_t)KT1 * NT * 32 * 16;

    zero_agg_kernel<<<(N_NODES * 4 + 255) / 256, 256, 0, stream>>>(agg);
    pack_weights_kernel<<<(KT1 * NT * 512 + KT2 * NT * 512 + 255) / 256, 256, 0, stream>>>(
        W1, W2, W1f, W2f);
    scatter_edges_kernel<<<(N_EDGES + 255) / 256, 256, 0, stream>>>(ei, ea, agg);
    mlp_kernel<<<N_NODES / BM, 256, 0, stream>>>(x, agg, W1f, W2f, b1, b2, out);
}
